// InternVLAA1_5171140624950
// MI455X (gfx1250) — matrix operand order repaired and checked by reading, not compiled, not run on hardware
//
#include <hip/hip_runtime.h>

// ---------------------------------------------------------------------------
// InternVL-style block for MI455X (gfx1250).
//  - all GEMMs / attention einsums on v_wmma_f32_16x16x32_bf16
//  - weights converted+transposed to bf16 [N,K] once per launch, so every
//    LDS tile (A, B0, B1, K, V) is a strided-2D -> contiguous copy staged by
//    the Tensor Data Mover (tensor_load_to_lds + s_wait_tensorcnt)
// ---------------------------------------------------------------------------

#define DEV __device__ __forceinline__

typedef __attribute__((ext_vector_type(16))) __bf16 v16bf;
typedef __attribute__((ext_vector_type(8)))  __bf16 v8bf;
typedef __attribute__((ext_vector_type(8)))  float  v8f;
typedef __attribute__((ext_vector_type(4)))  unsigned int v4u;
typedef __attribute__((ext_vector_type(8)))  int v8i;
typedef __attribute__((ext_vector_type(4)))  int v4i;

// model constants (fixed by the reference)
#define B_     2
#define SUND_  1536
#define SGEN_  256
#define SACT_  256
#define S_     2048
#define H_     16
#define KV_    8
#define D_     128
#define SCALE_ 0.08838834764831845f   // 128^-0.5

#if defined(__gfx1250__) && __has_builtin(__builtin_amdgcn_tensor_load_to_lds) && \
    __has_builtin(__builtin_amdgcn_s_wait_tensorcnt)
#define USE_TDM 1
#else
#define USE_TDM 0
#endif

DEV v8f zero8() { v8f z = {0.f,0.f,0.f,0.f,0.f,0.f,0.f,0.f}; return z; }

#if USE_TDM
// 2D tile DMA: global (row pitch stride_elems, element = 2 bytes) -> LDS
// contiguous tile of tile_d0 x tile_d1 elements.  EXEC is ignored by TENSOR
// ops; caller issues from one wave, then s_wait_tensorcnt + barrier.
DEV void tdm_load_2d(unsigned lds_off, const void* gptr,
                     unsigned tile_d0, unsigned tile_d1, unsigned stride_elems) {
  const unsigned long long ga = (unsigned long long)gptr;
  v4u g0;
  g0[0] = 1u;                                    // count=1, user descriptor
  g0[1] = lds_off;                               // lds_addr (bytes)
  g0[2] = (unsigned)ga;                          // global_addr[31:0]
  g0[3] = ((unsigned)(ga >> 32) & 0x01FFFFFFu)   // global_addr[56:32]
          | (2u << 30);                          // type = 2 ("image")
  v8i g1;
  g1[0] = (int)(1u << 16);                       // data_size = 1 (2 bytes)
  g1[1] = (int)(tile_d0 << 16);                  // tensor_dim0[15:0]
  g1[2] = (int)(0xFFFFu << 16);                  // tensor_dim1[15:0] = big
  g1[3] = (int)(tile_d0 << 16);                  // tile_dim0 | tensor_dim1 hi=0
  g1[4] = (int)tile_d1;                          // tile_dim1 (tile_dim2 = 0)
  g1[5] = (int)stride_elems;                     // tensor_dim0_stride[31:0]
  g1[6] = 0;                                     // stride hi / dim1_stride lo
  g1[7] = 0;
  v4i z4 = {0, 0, 0, 0};
#if defined(__clang_major__) && (__clang_major__ >= 23)
  v8i z8 = {0, 0, 0, 0, 0, 0, 0, 0};
  __builtin_amdgcn_tensor_load_to_lds(g0, g1, z4, z4, z8, 0);
#else
  __builtin_amdgcn_tensor_load_to_lds(g0, g1, z4, z4, 0);
#endif
}
#endif

// A-matrix 16x32 bf16 lane layout (matches B): element e -> K = half*16 + e
// i.e. 16 contiguous K values per lane.

DEV v16bf fragA_from_row(const __bf16* row, int half) {
  const v8bf a0 = *(const v8bf*)(row + half * 16);
  const v8bf a1 = *(const v8bf*)(row + half * 16 + 8);
  v16bf r;
#pragma unroll
  for (int i = 0; i < 8; ++i) { r[i] = a0[i]; r[i + 8] = a1[i]; }
  return r;
}

// B-matrix 32x16 bf16: lane col N = lane%16, element e -> K = half*16 + e
// => 16 contiguous K values per lane; caller passes 32B-aligned pointer.
DEV v16bf fragB_contig(const __bf16* p) { return *(const v16bf*)p; }

// ---------------------------------------------------------------------------
// Tiled transpose + convert: src f32 [K,N] -> dst bf16 [N,K].
// Grid (N/32, K/32), 256 threads; coalesced on both sides.
// ---------------------------------------------------------------------------
__global__ __launch_bounds__(256) void transpose_bf16_kernel(
    const float* __restrict__ src, __bf16* __restrict__ dst, int K, int N) {
  __shared__ float tile[32][33];
  const int tx = threadIdx.x & 31, ty = threadIdx.x >> 5;   // ty: 0..7
  const int nt = blockIdx.x * 32, kt = blockIdx.y * 32;
#pragma unroll
  for (int i = 0; i < 4; ++i)
    tile[ty + i * 8][tx] = src[(size_t)(kt + ty + i * 8) * N + nt + tx];
  __syncthreads();
#pragma unroll
  for (int i = 0; i < 4; ++i)
    dst[(size_t)(nt + ty + i * 8) * K + kt + tx] = (__bf16)tile[tx][ty + i * 8];
}

// ---------------------------------------------------------------------------
// V scatter: v_t f32 [T, KV*D] -> Vt bf16 [B, KV, D, S]  (d-major, s contig)
// so attention V tiles become TDM-loadable.  Grid (D/32, T/32, KV).
// ---------------------------------------------------------------------------
__global__ __launch_bounds__(256) void v_scatter_kernel(
    const float* __restrict__ v_t, __bf16* __restrict__ Vt,
    int S_str, int s_off) {
  __shared__ float tile[32][33];
  const int tx = threadIdx.x & 31, ty = threadIdx.x >> 5;
  const int d0 = blockIdx.x * 32, t0 = blockIdx.y * 32, kvh = blockIdx.z;
#pragma unroll
  for (int i = 0; i < 4; ++i)
    tile[ty + i * 8][tx] =
        v_t[(size_t)(t0 + ty + i * 8) * (KV_ * D_) + kvh * D_ + d0 + tx];
  __syncthreads();
  const int tok = t0 + tx;                   // 32-token tile: single stream/b
  const int b = tok / S_str, sg = s_off + tok % S_str;
#pragma unroll
  for (int i = 0; i < 4; ++i)
    Vt[((size_t)(b * KV_ + kvh) * D_ + d0 + ty + i * 8) * S_ + sg] =
        (__bf16)tile[tx][ty + i * 8];
}

// ---------------------------------------------------------------------------
// Row RMSNorm: f32 [rows, h] -> bf16 [rows, h].   One block per row.
// ---------------------------------------------------------------------------
__global__ __launch_bounds__(256) void rmsnorm_kernel(
    const float* __restrict__ x, const float* __restrict__ w,
    __bf16* __restrict__ out, int h) {
  const int row = blockIdx.x;
  const float* xr = x + (size_t)row * h;
  float ss = 0.f;
  for (int i = threadIdx.x; i < h; i += 256) { float v = xr[i]; ss += v * v; }
#pragma unroll
  for (int o = 16; o > 0; o >>= 1) ss += __shfl_xor(ss, o, 32);
  __shared__ float red[8];
  __shared__ float rs;
  if ((threadIdx.x & 31) == 0) red[threadIdx.x >> 5] = ss;
  __syncthreads();
  if (threadIdx.x == 0) {
    float t = 0.f;
#pragma unroll
    for (int i = 0; i < 8; ++i) t += red[i];
    rs = rsqrtf(t / (float)h + 1e-6f);
  }
  __syncthreads();
  const float rinv = rs;
  __bf16* orow = out + (size_t)row * h;
  for (int i = threadIdx.x; i < h; i += 256)
    orow[i] = (__bf16)(xr[i] * rinv * w[i]);
}

// ---------------------------------------------------------------------------
// Per-head RMSNorm + RoPE + scatter into [B, S, NH, D] bf16 (Q and K only).
// One wave per (token, head); lane owns d in {lane, lane+32, lane+64, lane+96}
// so the rotate partner (d +/- 64) is already resident in the lane.
// scale_q: multiply by D^-0.5 (fold softmax scale into Q).
// ---------------------------------------------------------------------------
__global__ __launch_bounds__(256) void qk_rope_kernel(
    const float* __restrict__ src, const float* __restrict__ nw,
    const int* __restrict__ posid, __bf16* __restrict__ dst,
    int T, int NH, int S_str, int s_off, int scale_q) {
  const int lane = threadIdx.x & 31;
  const int gw = blockIdx.x * 8 + (threadIdx.x >> 5);
  if (gw >= T * NH) return;
  const int tok = gw / NH, hd = gw % NH;
  const int b = tok / S_str, sl = tok % S_str;
  const int sg = s_off + sl;
  const float* x = src + (size_t)tok * (NH * D_) + hd * D_;
  float v0 = x[lane], v1 = x[lane + 32], v2 = x[lane + 64], v3 = x[lane + 96];
  float ss = v0 * v0 + v1 * v1 + v2 * v2 + v3 * v3;
#pragma unroll
  for (int o = 16; o > 0; o >>= 1) ss += __shfl_xor(ss, o, 32);
  const float rinv = rsqrtf(ss * (1.0f / 128.0f) + 1e-6f);
  v0 *= rinv * nw[lane];
  v1 *= rinv * nw[lane + 32];
  v2 *= rinv * nw[lane + 64];
  v3 *= rinv * nw[lane + 96];
  const float p = (float)posid[b * S_ + sg];
  const float inv0 = powf(1.0e6f, -(float)lane / 64.0f);
  const float inv1 = powf(1.0e6f, -(float)(lane + 32) / 64.0f);
  float s0, c0, s1, c1;
  sincosf(p * inv0, &s0, &c0);
  sincosf(p * inv1, &s1, &c1);
  float o0 = v0 * c0 - v2 * s0;
  float o2 = v2 * c0 + v0 * s0;
  float o1 = v1 * c1 - v3 * s1;
  float o3 = v3 * c1 + v1 * s1;
  if (scale_q) { o0 *= SCALE_; o1 *= SCALE_; o2 *= SCALE_; o3 *= SCALE_; }
  __bf16* o = dst + ((size_t)(b * S_ + sg) * NH + hd) * D_;
  o[lane] = (__bf16)o0; o[lane + 32] = (__bf16)o1;
  o[lane + 64] = (__bf16)o2; o[lane + 96] = (__bf16)o3;
}

// ---------------------------------------------------------------------------
// bf16 WMMA GEMM: C[M,N] = A[M,K](bf16) @ W (bf16, pre-transposed [N,K]).
// Block tile 128x128, K-slab 32, 8 waves (4x2), wave tile 32x64 (8 WMMA/slab).
// All LDS tiles share one shape (128 rows x 32 contiguous bf16, pitch K) and
// are staged by the Tensor Data Mover when available.
// EP==0: store f32.  EP==1: store f32 + residual R.
// DUAL: two weights, epilogue silu(acc0)*acc1 -> bf16 (SwiGLU).
// M % 128 == 0, N % 128 == 0, K % 32 == 0 hold for every call here.
// ---------------------------------------------------------------------------
template <int EP, bool DUAL>
__global__ __launch_bounds__(256) void gemm_kernel(
    const __bf16* __restrict__ A, const __bf16* __restrict__ W0t,
    const __bf16* __restrict__ W1t, const float* __restrict__ R,
    float* __restrict__ Cf, __bf16* __restrict__ Cb,
    int M, int N, int K) {
  __shared__ __align__(64) __bf16 sA[128][32];
  __shared__ __align__(64) __bf16 sB0[128][32];             // [n][k]
  __shared__ __align__(64) __bf16 sB1[DUAL ? 128 : 1][32];

  const int tid = threadIdx.x;
  const int lane = tid & 31, wid = tid >> 5;
  const int half = lane >> 4, l16 = lane & 15;
  const int bm = blockIdx.y * 128;
  const int bn = blockIdx.x * 128;
  const int wm = (wid >> 1) * 32;
  const int wn = (wid & 1) * 64;

  v8f acc0[2][4], acc1[2][4];
#pragma unroll
  for (int a = 0; a < 2; ++a)
#pragma unroll
    for (int c = 0; c < 4; ++c) { acc0[a][c] = zero8(); acc1[a][c] = zero8(); }

  const int nkb = K >> 5;
  for (int kb = 0; kb < nkb; ++kb) {
#if USE_TDM
    if (wid == 0) {   // DMA A, B0 (and B1): 128 rows x 32 bf16, row pitch K
      tdm_load_2d((unsigned)(unsigned long long)&sA[0][0],
                  A + (size_t)bm * K + kb * 32, 32u, 128u, (unsigned)K);
      tdm_load_2d((unsigned)(unsigned long long)&sB0[0][0],
                  W0t + (size_t)bn * K + kb * 32, 32u, 128u, (unsigned)K);
      if (DUAL)
        tdm_load_2d((unsigned)(unsigned long long)&sB1[0][0],
                    W1t + (size_t)bn * K + kb * 32, 32u, 128u, (unsigned)K);
      __builtin_amdgcn_s_wait_tensorcnt(0);
    }
#else
    { // manual staging: every tile is 2x16B contiguous per thread
      const int row = tid >> 1, col = (tid & 1) * 16;
      const __bf16* ap = A + (size_t)(bm + row) * K + kb * 32 + col;
      *(uint4*)&sA[row][col] = *(const uint4*)ap;
      *(uint4*)&sA[row][col + 8] = *(const uint4*)(ap + 8);
      const __bf16* bp = W0t + (size_t)(bn + row) * K + kb * 32 + col;
      *(uint4*)&sB0[row][col] = *(const uint4*)bp;
      *(uint4*)&sB0[row][col + 8] = *(const uint4*)(bp + 8);
      if (kb + 1 < nkb) __builtin_prefetch(bp + 32, 0, 1);
      if (DUAL) {
        const __bf16* cp = W1t + (size_t)(bn + row) * K + kb * 32 + col;
        *(uint4*)&sB1[row][col] = *(const uint4*)cp;
        *(uint4*)&sB1[row][col + 8] = *(const uint4*)(cp + 8);
      }
    }
#endif
    __syncthreads();
    const v16bf af0 = fragA_from_row(&sA[wm + l16][0], half);
    const v16bf af1 = fragA_from_row(&sA[wm + 16 + l16][0], half);
#pragma unroll
    for (int c = 0; c < 4; ++c) {
      const v16bf b0 = fragB_contig(&sB0[wn + c * 16 + l16][half * 16]);
      acc0[0][c] = __builtin_amdgcn_wmma_f32_16x16x32_bf16(
          false, af0, false, b0, (short)0, acc0[0][c], false, false);
      acc0[1][c] = __builtin_amdgcn_wmma_f32_16x16x32_bf16(
          false, af1, false, b0, (short)0, acc0[1][c], false, false);
      if (DUAL) {
        const v16bf b1 = fragB_contig(&sB1[wn + c * 16 + l16][half * 16]);
        acc1[0][c] = __builtin_amdgcn_wmma_f32_16x16x32_bf16(
            false, af0, false, b1, (short)0, acc1[0][c], false, false);
        acc1[1][c] = __builtin_amdgcn_wmma_f32_16x16x32_bf16(
            false, af1, false, b1, (short)0, acc1[1][c], false, false);
      }
    }
    __syncthreads();
  }

  // epilogue (C layout: row = r + 8*half, col = lane%16)
#pragma unroll
  for (int a = 0; a < 2; ++a) {
#pragma unroll
    for (int c = 0; c < 4; ++c) {
#pragma unroll
      for (int r = 0; r < 8; ++r) {
        const int gr = bm + wm + a * 16 + r + 8 * half;
        const int gc = bn + wn + c * 16 + l16;
        const size_t idx = (size_t)gr * N + gc;
        if (DUAL) {
          const float g = acc0[a][c][r], u = acc1[a][c][r];
          const float s = g / (1.f + expf(-g));
          Cb[idx] = (__bf16)(s * u);
        } else if (EP == 1) {
          Cf[idx] = acc0[a][c][r] + R[idx];
        } else {
          Cf[idx] = acc0[a][c][r];
        }
      }
    }
  }
}

// ---------------------------------------------------------------------------
// Causal GQA flash attention, bf16 WMMA, online softmax.
// Block = (q-tile of 128 rows, head, batch); 8 waves x 16 q-rows.
// K tile [32 keys][128 d] and V tile [128 d][32 keys] both staged by TDM
// (V comes pre-transposed as Vt[B,KV,D,S]).  Scale folded into Q.
// ---------------------------------------------------------------------------
__global__ __launch_bounds__(256) void attn_kernel(
    const __bf16* __restrict__ Qb, const __bf16* __restrict__ Kb,
    const __bf16* __restrict__ Vt, __bf16* __restrict__ att_u,
    __bf16* __restrict__ att_g, __bf16* __restrict__ att_a) {
  __shared__ __align__(64) __bf16 sK[32][128];      // [key][d]
  __shared__ __align__(64) __bf16 sVt[128][32];     // [d][key]
  __shared__ __align__(64) __bf16 sP[8][16][32];    // per-wave P tile

  const int tid = threadIdx.x;
  const int lane = tid & 31, wid = tid >> 5;
  const int half = lane >> 4, l16 = lane & 15;
  const int head = blockIdx.y, b = blockIdx.z;
  const int kvh = head >> 1;                        // H/KV = 2
  const int qtile = blockIdx.x * 128;
  const int qrow0 = qtile + wid * 16;

  // Q fragments in registers (rows = qrow0 + lane%16, 4 K-slabs over D=128)
  v16bf qf[4];
  {
    const __bf16* qp = Qb + ((size_t)(b * S_ + qrow0 + l16) * H_ + head) * D_;
#pragma unroll
    for (int f = 0; f < 4; ++f) qf[f] = fragA_from_row(qp + f * 32, half);
  }

  v8f oacc[8];
  float mstate[8], lstate[8];
#pragma unroll
  for (int i = 0; i < 8; ++i) { oacc[i] = zero8(); mstate[i] = -3.0e38f; lstate[i] = 0.f; }

  const int wq_max = qrow0 + 15;
  const int nkb = (qtile >> 5) + 4;                 // causal coverage of tile

  for (int kb = 0; kb < nkb; ++kb) {
#if USE_TDM
    if (wid == 0) {
      // K tile: 32 keys x 128 bf16, row pitch KV*D
      tdm_load_2d((unsigned)(unsigned long long)&sK[0][0],
                  Kb + ((size_t)(b * S_ + kb * 32) * KV_ + kvh) * D_,
                  128u, 32u, (unsigned)(KV_ * D_));
      // V tile: 128 d-rows x 32 keys, row pitch S
      tdm_load_2d((unsigned)(unsigned long long)&sVt[0][0],
                  Vt + ((size_t)(b * KV_ + kvh) * D_) * S_ + kb * 32,
                  32u, 128u, (unsigned)S_);
      __builtin_amdgcn_s_wait_tensorcnt(0);
    }
#else
    { // stage K row-major
      const int key = tid >> 3, dseg = (tid & 7) * 16;
      const __bf16* kp =
          Kb + ((size_t)(b * S_ + kb * 32 + key) * KV_ + kvh) * D_ + dseg;
      *(uint4*)&sK[key][dseg] = *(const uint4*)kp;
      *(uint4*)&sK[key][dseg + 8] = *(const uint4*)(kp + 8);
    }
    { // stage V tile from pre-transposed Vt
      const int row = tid >> 1, col = (tid & 1) * 16;
      const __bf16* vp =
          Vt + ((size_t)(b * KV_ + kvh) * D_ + row) * S_ + kb * 32 + col;
      *(uint4*)&sVt[row][col] = *(const uint4*)vp;
      *(uint4*)&sVt[row][col + 8] = *(const uint4*)(vp + 8);
    }
#endif
    __syncthreads();
    if (kb * 32 <= wq_max) {
      // scores S = Q @ K^T : 16x32 in two 16x16 frags, K-dim = D in 4 slabs
      v8f sc[2] = { zero8(), zero8() };
#pragma unroll
      for (int c = 0; c < 2; ++c)
#pragma unroll
        for (int f = 0; f < 4; ++f) {
          const v16bf bk = fragB_contig(&sK[c * 16 + l16][f * 32 + half * 16]);
          sc[c] = __builtin_amdgcn_wmma_f32_16x16x32_bf16(
              false, qf[f], false, bk, (short)0, sc[c], false, false);
        }
      // causal mask
#pragma unroll
      for (int c = 0; c < 2; ++c) {
        const int kpos = kb * 32 + c * 16 + l16;
#pragma unroll
        for (int r = 0; r < 8; ++r)
          if (kpos > qrow0 + r + 8 * half) sc[c][r] = -3.0e38f;
      }
      // online softmax: row reductions across 16-lane halves
#pragma unroll
      for (int r = 0; r < 8; ++r) {
        float mb = fmaxf(sc[0][r], sc[1][r]);
#pragma unroll
        for (int o = 8; o > 0; o >>= 1) mb = fmaxf(mb, __shfl_xor(mb, o, 16));
        const float mnew = fmaxf(mstate[r], mb);
        const float fac = expf(mstate[r] - mnew);
        mstate[r] = mnew;
        const float p0 = expf(sc[0][r] - mnew);
        const float p1 = expf(sc[1][r] - mnew);
        float rsum = p0 + p1;
#pragma unroll
        for (int o = 8; o > 0; o >>= 1) rsum += __shfl_xor(rsum, o, 16);
        lstate[r] = lstate[r] * fac + rsum;
#pragma unroll
        for (int oc = 0; oc < 8; ++oc) oacc[oc][r] *= fac;
        const int m = r + 8 * half;                 // C-layout row -> P row
        sP[wid][m][l16] = (__bf16)p0;
        sP[wid][m][16 + l16] = (__bf16)p1;
      }
      // P @ V  (same-wave LDS ops are in-order: stores above are visible)
      const v16bf pf = fragA_from_row(&sP[wid][l16][0], half);
#pragma unroll
      for (int oc = 0; oc < 8; ++oc) {
        const v16bf bv = fragB_contig(&sVt[oc * 16 + l16][half * 16]);
        oacc[oc] = __builtin_amdgcn_wmma_f32_16x16x32_bf16(
            false, pf, false, bv, (short)0, oacc[oc], false, false);
      }
    }
    __syncthreads();
  }

  // epilogue: normalize and scatter to per-stream att buffers [T_str, H*D]
#pragma unroll
  for (int oc = 0; oc < 8; ++oc) {
#pragma unroll
    for (int r = 0; r < 8; ++r) {
      const int s = qrow0 + r + 8 * half;
      const int col = head * D_ + oc * 16 + l16;
      const float val = oacc[oc][r] / lstate[r];
      __bf16* dst; int t;
      if (s < SUND_)               { dst = att_u; t = b * SUND_ + s; }
      else if (s < SUND_ + SGEN_)  { dst = att_g; t = b * SGEN_ + (s - SUND_); }
      else                         { dst = att_a; t = b * SACT_ + (s - SUND_ - SGEN_); }
      dst[(size_t)t * (H_ * D_) + col] = (__bf16)val;
    }
  }
}

// ---------------------------------------------------------------------------
// Host orchestration
// ---------------------------------------------------------------------------
extern "C" void kernel_launch(void* const* d_in, const int* in_sizes, int n_in,
                              void* d_out, int out_size, void* d_ws, size_t ws_size,
                              hipStream_t stream) {
  (void)in_sizes; (void)n_in; (void)out_size; (void)ws_size;
  const float* hid[3] = { (const float*)d_in[0], (const float*)d_in[1],
                          (const float*)d_in[2] };
  const int* posid = (const int*)d_in[36];

  const int Tt[3]   = { B_ * SUND_, B_ * SGEN_, B_ * SACT_ };
  const int Hh[3]   = { 2048, 1024, 1024 };
  const int Ii[3]   = { 6144, 3072, 3072 };
  const int soff[3] = { 0, SUND_, SUND_ + SGEN_ };
  const int sstr[3] = { SUND_, SGEN_, SACT_ };

  // params: 0 ln_in 1 wq 2 wk 3 wv 4 qn 5 kn 6 wo 7 ln_post 8 wg 9 wu 10 wd
  auto prm = [&](int s, int i) -> const float* {
    return (const float*)d_in[3 + 11 * s + i];
  };

  // workspace carve (all offsets 256B aligned)
  char* ws = (char*)d_ws;
  size_t off = 0;
  auto carve = [&](size_t bytes) -> char* {
    char* p = ws + off; off += (bytes + 255) & ~(size_t)255; return p;
  };
  __bf16* h_bf = (__bf16*)carve((size_t)3072 * 2048 * 2);     // norm out / h2
  float*  q_t  = (float*)carve((size_t)3072 * 2048 * 4);
  float*  k_t  = (float*)carve((size_t)3072 * 1024 * 4);
  float*  v_t  = (float*)carve((size_t)3072 * 1024 * 4);
  __bf16* Qb   = (__bf16*)carve((size_t)B_ * S_ * H_ * D_ * 2);
  __bf16* Kb   = (__bf16*)carve((size_t)B_ * S_ * KV_ * D_ * 2);
  __bf16* Vt   = (__bf16*)carve((size_t)B_ * KV_ * D_ * S_ * 2);
  __bf16* att_s[3];
  att_s[0] = (__bf16*)carve((size_t)B_ * SUND_ * 2048 * 2);
  att_s[1] = (__bf16*)carve((size_t)B_ * SGEN_ * 2048 * 2);
  att_s[2] = (__bf16*)carve((size_t)B_ * SACT_ * 2048 * 2);
  float*  o_f  = (float*)carve((size_t)3072 * 2048 * 4);
  __bf16* a_bf = (__bf16*)carve((size_t)3072 * 6144 * 2);
  // bf16 pre-transposed weights (sized for the largest stream, reused)
  __bf16* wq_bf = (__bf16*)carve((size_t)2048 * 2048 * 2);
  __bf16* wk_bf = (__bf16*)carve((size_t)2048 * 1024 * 2);
  __bf16* wv_bf = (__bf16*)carve((size_t)2048 * 1024 * 2);
  __bf16* wo_bf = (__bf16*)carve((size_t)2048 * 2048 * 2);
  __bf16* wg_bf = (__bf16*)carve((size_t)2048 * 6144 * 2);
  __bf16* wu_bf = (__bf16*)carve((size_t)2048 * 6144 * 2);
  __bf16* wd_bf = (__bf16*)carve((size_t)6144 * 2048 * 2);

  float* outp[3];
  outp[0] = (float*)d_out;
  outp[1] = outp[0] + (size_t)B_ * SUND_ * 2048;
  outp[2] = outp[1] + (size_t)B_ * SGEN_ * 1024;

  // ---- phase A: norms, QKV projections, rope, KV-cache scatter ----
  for (int s3 = 0; s3 < 3; ++s3) {
    const int T = Tt[s3], h = Hh[s3];
    transpose_bf16_kernel<<<dim3(2048 / 32, h / 32), 256, 0, stream>>>(
        prm(s3, 1), wq_bf, h, 2048);
    transpose_bf16_kernel<<<dim3(1024 / 32, h / 32), 256, 0, stream>>>(
        prm(s3, 2), wk_bf, h, 1024);
    transpose_bf16_kernel<<<dim3(1024 / 32, h / 32), 256, 0, stream>>>(
        prm(s3, 3), wv_bf, h, 1024);
    rmsnorm_kernel<<<T, 256, 0, stream>>>(hid[s3], prm(s3, 0), h_bf, h);
    gemm_kernel<0, false><<<dim3(2048 / 128, T / 128), 256, 0, stream>>>(
        h_bf, wq_bf, nullptr, nullptr, q_t, nullptr, T, 2048, h);
    gemm_kernel<0, false><<<dim3(1024 / 128, T / 128), 256, 0, stream>>>(
        h_bf, wk_bf, nullptr, nullptr, k_t, nullptr, T, 1024, h);
    gemm_kernel<0, false><<<dim3(1024 / 128, T / 128), 256, 0, stream>>>(
        h_bf, wv_bf, nullptr, nullptr, v_t, nullptr, T, 1024, h);
    qk_rope_kernel<<<(T * H_) / 8, 256, 0, stream>>>(
        q_t, prm(s3, 4), posid, Qb, T, H_, sstr[s3], soff[s3], 1);
    qk_rope_kernel<<<(T * KV_) / 8, 256, 0, stream>>>(
        k_t, prm(s3, 5), posid, Kb, T, KV_, sstr[s3], soff[s3], 0);
    v_scatter_kernel<<<dim3(D_ / 32, T / 32, KV_), 256, 0, stream>>>(
        v_t, Vt, sstr[s3], soff[s3]);
  }

  // ---- attention over the full concatenated sequence ----
  attn_kernel<<<dim3(S_ / 128, H_, B_), 256, 0, stream>>>(
      Qb, Kb, Vt, att_s[0], att_s[1], att_s[2]);

  // ---- phase B: o-proj + residual, post-norm, SwiGLU MLP, down-proj ----
  for (int s3 = 0; s3 < 3; ++s3) {
    const int T = Tt[s3], h = Hh[s3], I = Ii[s3];
    transpose_bf16_kernel<<<dim3(h / 32, 2048 / 32), 256, 0, stream>>>(
        prm(s3, 6), wo_bf, 2048, h);
    transpose_bf16_kernel<<<dim3(I / 32, h / 32), 256, 0, stream>>>(
        prm(s3, 8), wg_bf, h, I);
    transpose_bf16_kernel<<<dim3(I / 32, h / 32), 256, 0, stream>>>(
        prm(s3, 9), wu_bf, h, I);
    transpose_bf16_kernel<<<dim3(h / 32, I / 32), 256, 0, stream>>>(
        prm(s3, 10), wd_bf, I, h);
    gemm_kernel<1, false><<<dim3(h / 128, T / 128), 256, 0, stream>>>(
        att_s[s3], wo_bf, nullptr, hid[s3], o_f, nullptr, T, h, 2048);
    rmsnorm_kernel<<<T, 256, 0, stream>>>(o_f, prm(s3, 7), h_bf, h);
    gemm_kernel<0, true><<<dim3(I / 128, T / 128), 256, 0, stream>>>(
        h_bf, wg_bf, wu_bf, nullptr, nullptr, a_bf, T, I, h);
    gemm_kernel<1, false><<<dim3(h / 128, T / 128), 256, 0, stream>>>(
        a_bf, wd_bf, nullptr, o_f, outp[s3], nullptr, T, h, I);
  }
}
